// TransformerEncoder_59296318488853
// MI455X (gfx1250) — matrix operand order repaired and checked by reading, not compiled, not run on hardware
//
#include <hip/hip_runtime.h>
#include <hip/hip_bf16.h>

typedef __attribute__((ext_vector_type(16))) _Float16 v16h;
typedef __attribute__((ext_vector_type(8)))  _Float16 h8;
typedef __attribute__((ext_vector_type(4)))  _Float16 h4;
typedef __attribute__((ext_vector_type(8)))  float    v8f;
typedef __attribute__((ext_vector_type(4)))  int      v4i;

#if defined(__gfx1250__) && __has_builtin(__builtin_amdgcn_global_load_async_to_lds_b128)
#define USE_ASYNC 1
#else
#define USE_ASYNC 0
#endif

namespace {
constexpr int S = 1024, D = 1024, H = 16, F = 4096, L = 2;
constexpr int DH = 64, B = 4, BS = B * S;
constexpr int BM = 256, BN = 64, BK = 32;
constexpr int NGRP = BM / 16;                       // 16 A row-groups
constexpr size_t XSZ = (size_t)BS * D;              // 4,194,304
constexpr size_t HSZ = (size_t)BS * F;              // 16,777,216
constexpr size_t PSZ = (size_t)B * H * S * S;       // 67,108,864 / layer
} // namespace

// ---------------------------------------------------------------------------
// WMMA fragment order used here (contiguous k, wave32):
//   A 16x32 f16 : lane = (r&15) | (k>>4)<<4     ; e = k&15
//   B 32x16 f16 : lane = (n&15) | (k>>4)<<4     ; e = k&15
// A tiles go to LDS in fragment order via async DMA (ping-pong buffers);
// B for weight GEMMs is pre-packed f16 in global (L2-resident), read directly.
// ---------------------------------------------------------------------------

// 16B global -> LDS copy: async DMA on CDNA5 (ASYNCcnt), plain copy fallback.
__device__ __forceinline__ void cp16_async(const _Float16* g, _Float16* l) {
#if USE_ASYNC
  __builtin_amdgcn_global_load_async_to_lds_b128((v4i*)g, (v4i*)l, 0, 0);
#else
  *(h8*)l = *(const h8*)g;
#endif
}
__device__ __forceinline__ void async_wait0() {
#if USE_ASYNC
#if __has_builtin(__builtin_amdgcn_s_wait_asynccnt)
  __builtin_amdgcn_s_wait_asynccnt(0);
#else
  asm volatile("s_wait_asynccnt 0x0" ::: "memory");
#endif
#endif
}

// ---- A tile (256x32 f16) -> LDS fragment order: 4 x 16B chunks per thread --
__device__ __forceinline__ void asyncA_f16(const _Float16* __restrict__ A, int lda,
                                           int row0, int k0,
                                           _Float16* __restrict__ buf, int tid) {
#pragma unroll
  for (int j = 0; j < 4; ++j) {
    const int idx = j * 256 + tid;
    const int r = idx >> 2, c8 = (idx & 3) * 8;
    const int g = r >> 4;
    const int lane = (r & 15) | ((c8 >> 4) << 4);
    const int e = c8 & 15;                              // 0 or 8
    cp16_async(A + (size_t)(row0 + r) * lda + (k0 + c8),
               buf + ((size_t)(g * 32 + lane) * 16 + e));
  }
}

// ---- B tile (scores, B = K^T): 1 x 16B chunk per thread --------------------
__device__ __forceinline__ void asyncBt_f16(const _Float16* __restrict__ Ksrc, int ldb,
                                            int k0, int n0,
                                            _Float16* __restrict__ buf, int tid) {
  const int n = tid >> 2, k8 = (tid & 3) * 8;
  const int lane = (n & 15) | ((k8 >> 4) << 4);
  const int t = n >> 4, e = k8 & 15;
  cp16_async(Ksrc + (size_t)(n0 + n) * ldb + (k0 + k8),
             buf + ((size_t)(t * 32 + lane) * 16 + e));
}

// ---- A staging: f32 source (probs), batched loads then cvt+store -----------
__device__ __forceinline__ void loadA_f32(float4 fr[8], const float* __restrict__ A,
                                          int lda, int row0, int k0, int tid) {
#pragma unroll
  for (int j = 0; j < 8; ++j) {
    const int idx = j * 256 + tid;
    const int r = idx >> 3, c4 = (idx & 7) * 4;
    fr[j] = *(const float4*)(A + (size_t)(row0 + r) * lda + (k0 + c4));
  }
}
__device__ __forceinline__ void storeA_f32(_Float16* __restrict__ dst,
                                           const float4 fr[8], int tid) {
#pragma unroll
  for (int j = 0; j < 8; ++j) {
    const int idx = j * 256 + tid;
    const int r = idx >> 3, c4 = (idx & 7) * 4;
    const int g = r >> 4;
    const int lane = (r & 15) | ((c4 >> 4) << 4);
    const int e = c4 & 15;
    h4 hv = {(_Float16)fr[j].x, (_Float16)fr[j].y, (_Float16)fr[j].z, (_Float16)fr[j].w};
    *(h4*)(dst + ((size_t)(g * 32 + lane) * 16 + e)) = hv;
  }
}

// ---- B staging (ctx): v, f16, contiguous along n (scatter LDS writes) ------
__device__ __forceinline__ void loadBv_f16(h8& br, const _Float16* __restrict__ Vsrc,
                                           int ldb, int k0, int tid) {
  const int kk = tid >> 3, n8 = (tid & 7) * 8;
  br = *(const h8*)(Vsrc + (size_t)(k0 + kk) * ldb + n8);
}
__device__ __forceinline__ void storeBv_f16(_Float16* __restrict__ dst,
                                            const h8 br, int tid) {
  const int kk = tid >> 3, n8 = (tid & 7) * 8;
  const int lh = kk >> 4, e = kk & 15;
#pragma unroll
  for (int u = 0; u < 8; ++u) {
    const int nn = n8 + u;
    const int lane = (nn & 15) | (lh << 4);
    dst[(size_t)((nn >> 4) * 32 + lane) * 16 + e] = br[u];
  }
}

// ---- 8 WMMA per BK=32 slab: 2 row-groups x 4 col tiles per wave ------------
__device__ __forceinline__ void mma_slab(const _Float16* __restrict__ As_f,
                                         const _Float16* __restrict__ Bf,  // LDS or global
                                         v8f acc[2][4], int lane, int wave) {
  const v16h a0 = *(const v16h*)(As_f + (size_t)((wave * 2 + 0) * 32 + lane) * 16);
  const v16h a1 = *(const v16h*)(As_f + (size_t)((wave * 2 + 1) * 32 + lane) * 16);
  v16h bfr[4];
#pragma unroll
  for (int t = 0; t < 4; ++t)
    bfr[t] = *(const v16h*)(Bf + (size_t)(t * 32 + lane) * 16);
#pragma unroll
  for (int t = 0; t < 4; ++t) {
    acc[0][t] = __builtin_amdgcn_wmma_f32_16x16x32_f16(false, a0, false, bfr[t],
                                                       (short)0, acc[0][t], false, false);
    acc[1][t] = __builtin_amdgcn_wmma_f32_16x16x32_f16(false, a1, false, bfr[t],
                                                       (short)0, acc[1][t], false, false);
  }
}

// ---------------------------------------------------------------------------
// One-time weight repack: f32 row-major (KxN) -> f16 fragment-packed
// [slab=k/32][ntile=n/16][lane][e], so B fragments stream straight from L2.
// ---------------------------------------------------------------------------
__global__ void __launch_bounds__(256)
pack_w_kernel(const float* __restrict__ W, _Float16* __restrict__ Wp, int N) {
  const size_t o = (size_t)blockIdx.x * 256 + threadIdx.x;
  const int e = (int)(o & 15);
  const int lane = (int)((o >> 4) & 31);
  const size_t rest = o >> 9;
  const int ntiles = N >> 4;
  const int nt = (int)(rest % ntiles);
  const size_t slab = rest / ntiles;
  const int kk = e + ((lane >> 4) << 4);
  const int n = nt * 16 + (lane & 15);
  Wp[o] = (_Float16)W[(slab * 32 + kk) * (size_t)N + n];
}

// ---------------------------------------------------------------------------
// C[M,N] = A[M,K](f16) @ Wp(packed f16) + bias ; OutT = float or f16; opt ReLU
// ---------------------------------------------------------------------------
template <typename OutT, bool RELU>
__global__ void __launch_bounds__(256)
gemm_bias_kernel(const _Float16* __restrict__ A, const _Float16* __restrict__ Bp,
                 const float* __restrict__ bias, OutT* __restrict__ C,
                 int K, int N) {
  __shared__ _Float16 As_f[2][NGRP * 32 * 16];    // 2 x 16 KB ping-pong
  const int tid = threadIdx.x, lane = tid & 31, wave = tid >> 5;
  const int row0 = blockIdx.y * BM, n0 = blockIdx.x * BN;
  const v8f vz = {0.f, 0.f, 0.f, 0.f, 0.f, 0.f, 0.f, 0.f};
  v8f acc[2][4];
#pragma unroll
  for (int gi = 0; gi < 2; ++gi)
#pragma unroll
    for (int t = 0; t < 4; ++t) acc[gi][t] = vz;

  const int nslab = K >> 5;
  const size_t ntiles = (size_t)(N >> 4);
  asyncA_f16(A, K, row0, 0, As_f[0], tid);
  async_wait0();
  __syncthreads();
  for (int s = 0; s < nslab; ++s) {
    if (s + 1 < nslab) asyncA_f16(A, K, row0, (s + 1) << 5, As_f[(s + 1) & 1], tid);
    const _Float16* bb = Bp + ((size_t)s * ntiles + (n0 >> 4)) * 512;
    mma_slab(As_f[s & 1], bb, acc, lane, wave);
    async_wait0();
    __syncthreads();
  }
  const int lm = lane & 15, lh = lane >> 4;
#pragma unroll
  for (int gi = 0; gi < 2; ++gi)
#pragma unroll
    for (int t = 0; t < 4; ++t)
#pragma unroll
      for (int r = 0; r < 8; ++r) {
        const int row = row0 + (wave * 2 + gi) * 16 + lh * 8 + r;
        const int col = n0 + t * 16 + lm;
        float val = acc[gi][t][r] + bias[col];
        if (RELU) val = fmaxf(val, 0.f);
        C[(size_t)row * N + col] = (OutT)val;
      }
}

// ---------------------------------------------------------------------------
// scores[b,h,q,k] = (q.k)/8, masked -> d_out probs slab (f32)
// ---------------------------------------------------------------------------
__global__ void __launch_bounds__(256)
attn_scores_kernel(const _Float16* __restrict__ Q, const _Float16* __restrict__ Kc,
                   const unsigned char* __restrict__ mask, float* __restrict__ P) {
  __shared__ _Float16 As_f[2][NGRP * 32 * 16];
  __shared__ _Float16 Bs_f[2][4 * 32 * 16];
  const int tid = threadIdx.x, lane = tid & 31, wave = tid >> 5;
  const int z = blockIdx.z, b = z >> 4, h = z & 15;
  const _Float16* q = Q + (size_t)b * S * D + h * DH;
  const _Float16* k = Kc + (size_t)b * S * D + h * DH;
  const int row0 = blockIdx.y * BM, n0 = blockIdx.x * BN;
  const v8f vz = {0.f, 0.f, 0.f, 0.f, 0.f, 0.f, 0.f, 0.f};
  v8f acc[2][4];
#pragma unroll
  for (int gi = 0; gi < 2; ++gi)
#pragma unroll
    for (int t = 0; t < 4; ++t) acc[gi][t] = vz;

  asyncA_f16(q, D, row0, 0, As_f[0], tid);
  asyncBt_f16(k, D, 0, n0, Bs_f[0], tid);
  async_wait0();
  __syncthreads();
#pragma unroll
  for (int s = 0; s < 2; ++s) {                  // DH = 2 slabs
    if (s == 0) {
      asyncA_f16(q, D, row0, 32, As_f[1], tid);
      asyncBt_f16(k, D, 32, n0, Bs_f[1], tid);
    }
    mma_slab(As_f[s], Bs_f[s], acc, lane, wave);
    async_wait0();
    __syncthreads();
  }
  const int lm = lane & 15, lh = lane >> 4;
  float* Pz = P + (size_t)z * S * S;
  const unsigned char* mb = mask + (size_t)b * S * S;
#pragma unroll
  for (int gi = 0; gi < 2; ++gi)
#pragma unroll
    for (int t = 0; t < 4; ++t)
#pragma unroll
      for (int r = 0; r < 8; ++r) {
        const int row = row0 + (wave * 2 + gi) * 16 + lh * 8 + r;
        const int col = n0 + t * 16 + lm;
        float val = acc[gi][t][r] * 0.125f;
        if (mb[(size_t)row * S + col]) val = -1.0e9f;
        Pz[(size_t)row * S + col] = val;
      }
}

// ---------------------------------------------------------------------------
// ctx[b,q,h*DH+d] = sum_k p[b,h,q,k] * v[b,k,h*DH+d]  -> f16 output
// (A operand is fp32 probs -> manual staged path with conversion)
// ---------------------------------------------------------------------------
__global__ void __launch_bounds__(256)
attn_ctx_kernel(const float* __restrict__ P, const _Float16* __restrict__ Vc,
                _Float16* __restrict__ ctx) {
  __shared__ _Float16 As_f[2][NGRP * 32 * 16];
  __shared__ _Float16 Bs_f[2][4 * 32 * 16];
  const int tid = threadIdx.x, lane = tid & 31, wave = tid >> 5;
  const int z = blockIdx.z, b = z >> 4, h = z & 15;
  const float* p = P + (size_t)z * S * S;
  const _Float16* v = Vc + (size_t)b * S * D + h * DH;
  const int row0 = blockIdx.y * BM;
  const v8f vz = {0.f, 0.f, 0.f, 0.f, 0.f, 0.f, 0.f, 0.f};
  v8f acc[2][4];
#pragma unroll
  for (int gi = 0; gi < 2; ++gi)
#pragma unroll
    for (int t = 0; t < 4; ++t) acc[gi][t] = vz;

  float4 fr[8]; h8 br;
  loadA_f32(fr, p, S, row0, 0, tid);
  loadBv_f16(br, v, D, 0, tid);
  const int nslab = S >> 5;
  for (int s = 0; s < nslab; ++s) {
    storeA_f32(As_f[s & 1], fr, tid);
    storeBv_f16(Bs_f[s & 1], br, tid);
    __syncthreads();
    if (s + 1 < nslab) {
      loadA_f32(fr, p, S, row0, (s + 1) << 5, tid);
      loadBv_f16(br, v, D, (s + 1) << 5, tid);
    }
    mma_slab(As_f[s & 1], Bs_f[s & 1], acc, lane, wave);
  }
  const int lm = lane & 15, lh = lane >> 4;
#pragma unroll
  for (int gi = 0; gi < 2; ++gi)
#pragma unroll
    for (int t = 0; t < 4; ++t)
#pragma unroll
      for (int r = 0; r < 8; ++r) {
        const int row = row0 + (wave * 2 + gi) * 16 + lh * 8 + r;
        const int col = t * 16 + lm;
        ctx[(size_t)b * S * D + (size_t)row * D + h * DH + col] = (_Float16)acc[gi][t][r];
      }
}

// ---------------------------------------------------------------------------
__global__ void __launch_bounds__(256) softmax_kernel(float* __restrict__ P) {
  float* p = P + (size_t)blockIdx.x * S;
  __shared__ float sd[256];
  const int tid = threadIdx.x;
  float v[4];
  float m = -3.0e38f;
#pragma unroll
  for (int j = 0; j < 4; ++j) { v[j] = p[tid + j * 256]; m = fmaxf(m, v[j]); }
  sd[tid] = m; __syncthreads();
  for (int s2 = 128; s2 > 0; s2 >>= 1) {
    if (tid < s2) sd[tid] = fmaxf(sd[tid], sd[tid + s2]);
    __syncthreads();
  }
  m = sd[0]; __syncthreads();
  float sum = 0.f;
#pragma unroll
  for (int j = 0; j < 4; ++j) { v[j] = __expf(v[j] - m); sum += v[j]; }
  sd[tid] = sum; __syncthreads();
  for (int s2 = 128; s2 > 0; s2 >>= 1) {
    if (tid < s2) sd[tid] += sd[tid + s2];
    __syncthreads();
  }
  const float inv = 1.0f / sd[0];
#pragma unroll
  for (int j = 0; j < 4; ++j) p[tid + j * 256] = v[j] * inv;
}

// x = LayerNorm(x + y) * g + be ; also emits f16 mirror for downstream GEMMs
__global__ void __launch_bounds__(256)
add_ln_kernel(float* __restrict__ x, const float* __restrict__ y,
              const float* __restrict__ g, const float* __restrict__ be,
              _Float16* __restrict__ xh) {
  float* xr = x + (size_t)blockIdx.x * D;
  _Float16* hr = xh + (size_t)blockIdx.x * D;
  const float* yr = y + (size_t)blockIdx.x * D;
  __shared__ float sd[256];
  const int tid = threadIdx.x;
  float v[4];
  float sum = 0.f;
#pragma unroll
  for (int j = 0; j < 4; ++j) { v[j] = xr[tid + j * 256] + yr[tid + j * 256]; sum += v[j]; }
  sd[tid] = sum; __syncthreads();
  for (int s2 = 128; s2 > 0; s2 >>= 1) {
    if (tid < s2) sd[tid] += sd[tid + s2];
    __syncthreads();
  }
  const float mean = sd[0] * (1.0f / D); __syncthreads();
  float vs = 0.f;
#pragma unroll
  for (int j = 0; j < 4; ++j) { const float d = v[j] - mean; vs += d * d; }
  sd[tid] = vs; __syncthreads();
  for (int s2 = 128; s2 > 0; s2 >>= 1) {
    if (tid < s2) sd[tid] += sd[tid + s2];
    __syncthreads();
  }
  const float inv = rsqrtf(sd[0] * (1.0f / D) + 1e-5f);
#pragma unroll
  for (int j = 0; j < 4; ++j) {
    const int d = tid + j * 256;
    const float o = (v[j] - mean) * inv * g[d] + be[d];
    xr[d] = o;
    hr[d] = (_Float16)o;
  }
}

__global__ void __launch_bounds__(256)
embed_kernel(const int* __restrict__ tok, const float* __restrict__ emb,
             const float* __restrict__ pe, float* __restrict__ x,
             _Float16* __restrict__ xh) {
  const int row = blockIdx.x;
  const int s = row & (S - 1);
  const int t = tok[row];
#pragma unroll
  for (int j = 0; j < 4; ++j) {
    const int d = threadIdx.x + j * 256;
    const float o = emb[(size_t)t * D + d] * 32.0f + pe[(size_t)s * D + d];
    x[(size_t)row * D + d] = o;
    xh[(size_t)row * D + d] = (_Float16)o;
  }
}

// ---------------------------------------------------------------------------
extern "C" void kernel_launch(void* const* d_in, const int* in_sizes, int n_in,
                              void* d_out, int out_size, void* d_ws, size_t ws_size,
                              hipStream_t stream) {
  (void)in_sizes; (void)n_in; (void)out_size; (void)ws_size;
  const int*           enc  = (const int*)d_in[0];
  const unsigned char* mask = (const unsigned char*)d_in[1];
  const float* emb = (const float*)d_in[2];
  const float* pe  = (const float*)d_in[3];
  const float* Wq  = (const float*)d_in[4];
  const float* bq  = (const float*)d_in[5];
  const float* Wk  = (const float*)d_in[6];
  const float* bk  = (const float*)d_in[7];
  const float* Wv  = (const float*)d_in[8];
  const float* bv  = (const float*)d_in[9];
  const float* Wo  = (const float*)d_in[10];
  const float* bo  = (const float*)d_in[11];
  const float* g1  = (const float*)d_in[12];
  const float* be1 = (const float*)d_in[13];
  const float* W1  = (const float*)d_in[14];
  const float* b1  = (const float*)d_in[15];
  const float* W2  = (const float*)d_in[16];
  const float* b2  = (const float*)d_in[17];
  const float* g2  = (const float*)d_in[18];
  const float* be2 = (const float*)d_in[19];
  float* out = (float*)d_out;

  // ---- workspace layout ----
  float*    xs  = (float*)d_ws;              // BSxD f32
  float*    tmp = xs + XSZ;                  // BSxD f32
  _Float16* xh  = (_Float16*)(tmp + XSZ);    // BSxD f16 mirror of xs
  _Float16* qh  = xh + XSZ;
  _Float16* kh  = qh + XSZ;
  _Float16* vh  = kh + XSZ;
  _Float16* ch  = vh + XSZ;                  // ctx f16
  _Float16* hb  = ch + XSZ;                  // BSxF f16
  _Float16* wq_p = hb + HSZ;                 // packed weights (f16, frag order)
  _Float16* wk_p = wq_p + (size_t)L * D * D;
  _Float16* wv_p = wk_p + (size_t)L * D * D;
  _Float16* wo_p = wv_p + (size_t)L * D * D;
  _Float16* w1_p = wo_p + (size_t)L * D * D;
  _Float16* w2_p = w1_p + (size_t)L * D * F;

  // ---- one-time weight repack into L2-friendly fragment-packed f16 ----
  for (int i = 0; i < L; ++i) {
    const size_t odd = (size_t)i * D * D, odf = (size_t)i * D * F;
    pack_w_kernel<<<(D * D) / 256, 256, 0, stream>>>(Wq + odd, wq_p + odd, D);
    pack_w_kernel<<<(D * D) / 256, 256, 0, stream>>>(Wk + odd, wk_p + odd, D);
    pack_w_kernel<<<(D * D) / 256, 256, 0, stream>>>(Wv + odd, wv_p + odd, D);
    pack_w_kernel<<<(D * D) / 256, 256, 0, stream>>>(Wo + odd, wo_p + odd, D);
    pack_w_kernel<<<(D * F) / 256, 256, 0, stream>>>(W1 + odf, w1_p + odf, F);
    pack_w_kernel<<<(F * D) / 256, 256, 0, stream>>>(W2 + odf, w2_p + odf, D);
  }

  embed_kernel<<<BS, 256, 0, stream>>>(enc, emb, pe, xs, xh);

  for (int i = 0; i < L; ++i) {
    const size_t odd = (size_t)i * D * D, odf = (size_t)i * D * F;
    const dim3 gD(D / BN, BS / BM);   // 16 x 16

    gemm_bias_kernel<_Float16, false><<<gD, 256, 0, stream>>>(xh, wq_p + odd, bq + i * D, qh, D, D);
    gemm_bias_kernel<_Float16, false><<<gD, 256, 0, stream>>>(xh, wk_p + odd, bk + i * D, kh, D, D);
    gemm_bias_kernel<_Float16, false><<<gD, 256, 0, stream>>>(xh, wv_p + odd, bv + i * D, vh, D, D);

    float* P = out + XSZ + (size_t)i * PSZ;
    attn_scores_kernel<<<dim3(S / BN, S / BM, B * H), 256, 0, stream>>>(qh, kh, mask, P);
    softmax_kernel<<<B * H * S, 256, 0, stream>>>(P);
    attn_ctx_kernel<<<dim3(1, S / BM, B * H), 256, 0, stream>>>(P, vh, ch);

    gemm_bias_kernel<float, false><<<gD, 256, 0, stream>>>(ch, wo_p + odd, bo + i * D, tmp, D, D);
    add_ln_kernel<<<BS, 256, 0, stream>>>(xs, tmp, g1 + i * D, be1 + i * D, xh);

    gemm_bias_kernel<_Float16, true><<<dim3(F / BN, BS / BM), 256, 0, stream>>>(xh, w1_p + odf, b1 + i * F, hb, D, F);
    gemm_bias_kernel<float, false><<<gD, 256, 0, stream>>>(hb, w2_p + odf, b2 + i * D, tmp, F, D);
    add_ln_kernel<<<BS, 256, 0, stream>>>(xs, tmp, g2 + i * D, be2 + i * D, xh);
  }

  (void)hipMemcpyAsync(out, xs, XSZ * sizeof(float), hipMemcpyDeviceToDevice, stream);
}
